// ConvLSTM_61692910240037
// MI455X (gfx1250) — compile-verified, matrix operand order checked by reading, not run on hardware
//
#include <hip/hip_runtime.h>

// ---------------------------------------------------------------------------
// Types for CDNA5 WMMA (wave32, 16x16x32 f16 -> f32)
// ---------------------------------------------------------------------------
typedef __attribute__((ext_vector_type(16))) _Float16 v16h;
typedef __attribute__((ext_vector_type(8)))  _Float16 v8h;
typedef __attribute__((ext_vector_type(8)))  float    v8f;

#define EPS 1e-5f

// ---------------------------------------------------------------------------
// Block-wide sum reduction (blockDim.x == 256)
// ---------------------------------------------------------------------------
__device__ __forceinline__ float block_reduce_sum(float v, float* red) {
    const int t = threadIdx.x;
    red[t] = v;
    __syncthreads();
    for (int off = 128; off > 0; off >>= 1) {
        if (t < off) red[t] += red[t + off];
        __syncthreads();
    }
    float r = red[0];
    __syncthreads();
    return r;
}

__device__ __forceinline__ float sigmoidf_(float x) { return 1.0f / (1.0f + expf(-x)); }

__device__ __forceinline__ v16h combine8(v8h lo, v8h hi) {
    return __builtin_shufflevector(lo, hi, 0,1,2,3,4,5,6,7,8,9,10,11,12,13,14,15);
}

// ---------------------------------------------------------------------------
// Generic GEMM: Y(M,N) = X(M,K) * W(N,K)^T + bias, f16 inputs, f32 accum.
// Block = 256 threads = 8 waves. Register-blocked: each wave owns a 32x64
// output patch (2 M-tiles x 4 N-tiles = 8 WMMA accumulators); block tile is
// 64 x 256. Per 32-K step a wave issues 12 b128 loads for 8 WMMAs, and A
// fragments are reused across the 4 N-tiles in registers.
// Optional f32 output (ldyf) and/or f16 output (ldyh, column offset ycol) --
// the f16 path writes straight into the next GEMM's concat input buffer.
// Requires: M % 64 == 0, N % 256 == 0, K % 32 == 0, 16B-aligned rows.
// ---------------------------------------------------------------------------
__global__ __launch_bounds__(256)
void gemm_f16_wmma(const _Float16* __restrict__ A, int lda,
                   const _Float16* __restrict__ W,
                   const float* __restrict__ bias,
                   float* __restrict__ Yf, int ldyf,
                   _Float16* __restrict__ Yh, int ldyh, int ycol,
                   int K)
{
    const int wave = threadIdx.x >> 5;
    const int lane = threadIdx.x & 31;
    const int hf   = lane >> 4;      // lane half: 0 or 1
    const int lm   = lane & 15;
    const int mBase = blockIdx.x * 64 + (wave & 1) * 32;    // wave owns 32 rows
    const int nBase = blockIdx.y * 256 + (wave >> 1) * 64;  // wave owns 64 cols

    const _Float16* __restrict__ arow0 = A + (size_t)(mBase + lm)      * (size_t)lda;
    const _Float16* __restrict__ arow1 = A + (size_t)(mBase + 16 + lm) * (size_t)lda;
    const _Float16* __restrict__ wrow0 = W + (size_t)(nBase + lm)      * (size_t)K;
    const _Float16* __restrict__ wrow1 = W + (size_t)(nBase + 16 + lm) * (size_t)K;
    const _Float16* __restrict__ wrow2 = W + (size_t)(nBase + 32 + lm) * (size_t)K;
    const _Float16* __restrict__ wrow3 = W + (size_t)(nBase + 48 + lm) * (size_t)K;

    v8f acc[8];
#pragma unroll
    for (int i = 0; i < 8; ++i) acc[i] = (v8f){0.f,0.f,0.f,0.f,0.f,0.f,0.f,0.f};

    for (int k0 = 0; k0 < K; k0 += 32) {
        const int ka = k0 + hf * 8;       // per-lane K runs of 8 halfs (ISA layout)
        v16h a0 = combine8(*(const v8h*)(arow0 + ka), *(const v8h*)(arow0 + ka + 16));
        v16h a1 = combine8(*(const v8h*)(arow1 + ka), *(const v8h*)(arow1 + ka + 16));
        v16h b0 = combine8(*(const v8h*)(wrow0 + ka), *(const v8h*)(wrow0 + ka + 16));
        v16h b1 = combine8(*(const v8h*)(wrow1 + ka), *(const v8h*)(wrow1 + ka + 16));
        v16h b2 = combine8(*(const v8h*)(wrow2 + ka), *(const v8h*)(wrow2 + ka + 16));
        v16h b3 = combine8(*(const v8h*)(wrow3 + ka), *(const v8h*)(wrow3 + ka + 16));
        acc[0] = __builtin_amdgcn_wmma_f32_16x16x32_f16(false, a0, false, b0, (short)0, acc[0], false, false);
        acc[1] = __builtin_amdgcn_wmma_f32_16x16x32_f16(false, a1, false, b0, (short)0, acc[1], false, false);
        acc[2] = __builtin_amdgcn_wmma_f32_16x16x32_f16(false, a0, false, b1, (short)0, acc[2], false, false);
        acc[3] = __builtin_amdgcn_wmma_f32_16x16x32_f16(false, a1, false, b1, (short)0, acc[3], false, false);
        acc[4] = __builtin_amdgcn_wmma_f32_16x16x32_f16(false, a0, false, b2, (short)0, acc[4], false, false);
        acc[5] = __builtin_amdgcn_wmma_f32_16x16x32_f16(false, a1, false, b2, (short)0, acc[5], false, false);
        acc[6] = __builtin_amdgcn_wmma_f32_16x16x32_f16(false, a0, false, b3, (short)0, acc[6], false, false);
        acc[7] = __builtin_amdgcn_wmma_f32_16x16x32_f16(false, a1, false, b3, (short)0, acc[7], false, false);
    }

#pragma unroll
    for (int nt = 0; nt < 4; ++nt) {
        const int n = nBase + nt * 16 + lm;
        const float bv = bias[n];
#pragma unroll
        for (int mt = 0; mt < 2; ++mt) {
            const v8f a = acc[nt * 2 + mt];
#pragma unroll
            for (int r = 0; r < 8; ++r) {
                const int m = mBase + mt * 16 + hf * 8 + r;  // C/D: VGPR r, lane half -> M+8
                float y = a[r] + bv;
                if (Yf) Yf[(size_t)m * (size_t)ldyf + n] = y;
                if (Yh) Yh[(size_t)m * (size_t)ldyh + (size_t)(ycol + n)] = (_Float16)y;
            }
        }
    }
}

// ---------------------------------------------------------------------------
// Weight packing: f32 (N,K) -> f16 (N,Kp) with zero pad (Kp >= K)
// ---------------------------------------------------------------------------
__global__ void pack_w_f16(const float* __restrict__ src, _Float16* __restrict__ dst,
                           int N, int K, int Kp)
{
    int idx = blockIdx.x * 256 + threadIdx.x;
    if (idx >= N * Kp) return;
    int n = idx / Kp, k = idx % Kp;
    dst[idx] = (k < K) ? (_Float16)src[n * K + k] : (_Float16)0.f;
}

// LSTM: pack [w_ih | w_hh] -> (1024,512) f16 and combined bias
__global__ void pack_gates(const float* __restrict__ wih, const float* __restrict__ whh,
                           const float* __restrict__ bih, const float* __restrict__ bhh,
                           _Float16* __restrict__ wg, float* __restrict__ bg)
{
    int idx = blockIdx.x * 256 + threadIdx.x;
    if (idx < 1024 * 512) {
        int r = idx / 512, k = idx % 512;
        wg[idx] = (_Float16)((k < 256) ? wih[r * 256 + k] : whh[r * 256 + (k - 256)]);
    }
    if (idx < 1024) bg[idx] = bih[idx] + bhh[idx];
}

// ---------------------------------------------------------------------------
// f32 -> f16 copy into a (possibly strided / column-offset) destination
// ---------------------------------------------------------------------------
__global__ void f32_to_f16_strided(const float* __restrict__ src, _Float16* __restrict__ dst,
                                   long long total, int cols, int dstStride, int dstOff)
{
    long long idx = (long long)blockIdx.x * 256 + threadIdx.x;
    if (idx >= total) return;
    long long r = idx / cols;
    int c = (int)(idx % cols);
    dst[r * (long long)dstStride + dstOff + c] = (_Float16)src[idx];
}

__global__ void f16_fill_zero(_Float16* __restrict__ dst, long long total, int cols,
                              int dstStride, int dstOff)
{
    long long idx = (long long)blockIdx.x * 256 + threadIdx.x;
    if (idx >= total) return;
    long long r = idx / cols;
    int c = (int)(idx % cols);
    dst[r * (long long)dstStride + dstOff + c] = (_Float16)0.f;
}

// ---------------------------------------------------------------------------
// Conv1: 6->16, 3x3, stride 1, pad 1, 15x15 -> 15x15, + GroupNorm(4) + ReLU
// One block (256 thr) per sample; input & output staged in LDS.
// ---------------------------------------------------------------------------
__global__ __launch_bounds__(256)
void conv1_gn_relu(const float* __restrict__ x, const float* __restrict__ w,
                   const float* __restrict__ bias, const float* __restrict__ gg,
                   const float* __restrict__ gb, float* __restrict__ out)
{
    __shared__ float sin[6 * 225];
    __shared__ float sout[16 * 225];
    __shared__ float red[256];
    __shared__ float s_mean, s_rstd;

    const int b = blockIdx.x;
    const float* xin = x + (size_t)b * (6 * 225);
    for (int i = threadIdx.x; i < 6 * 225; i += 256) sin[i] = xin[i];
    __syncthreads();

    for (int o = threadIdx.x; o < 16 * 225; o += 256) {
        int c = o / 225, p = o % 225, oy = p / 15, ox = p % 15;
        float acc = bias[c];
        for (int ic = 0; ic < 6; ++ic) {
            const float* wc = w + ((c * 6 + ic) * 9);
            const float* si = sin + ic * 225;
#pragma unroll
            for (int ky = 0; ky < 3; ++ky) {
                int iy = oy + ky - 1;
                if (iy < 0 || iy >= 15) continue;
#pragma unroll
                for (int kx = 0; kx < 3; ++kx) {
                    int ix = ox + kx - 1;
                    if (ix < 0 || ix >= 15) continue;
                    acc += si[iy * 15 + ix] * wc[ky * 3 + kx];
                }
            }
        }
        sout[o] = acc;
    }
    __syncthreads();

    // GroupNorm: 4 groups of 4 channels * 225 = 900 elements
    for (int g = 0; g < 4; ++g) {
        float s = 0.f, s2 = 0.f;
        for (int i = threadIdx.x; i < 900; i += 256) {
            float v = sout[g * 900 + i];
            s += v; s2 += v * v;
        }
        float ts = block_reduce_sum(s, red);
        if (threadIdx.x == 0) s_mean = ts * (1.f / 900.f);
        __syncthreads();
        float ts2 = block_reduce_sum(s2, red);
        if (threadIdx.x == 0) {
            float m = s_mean;
            s_rstd = rsqrtf(ts2 * (1.f / 900.f) - m * m + EPS);
        }
        __syncthreads();
        float m = s_mean, rs = s_rstd;
        for (int i = threadIdx.x; i < 900; i += 256) {
            int c = g * 4 + i / 225;
            float v = (sout[g * 900 + i] - m) * rs * gg[c] + gb[c];
            out[(size_t)b * 3600 + g * 900 + i] = fmaxf(v, 0.f);
        }
        __syncthreads();
    }
}

// ---------------------------------------------------------------------------
// Conv2: 16->32, 3x3, stride 2, pad 1, 15x15 -> 8x8, + GroupNorm(8) + ReLU
// ---------------------------------------------------------------------------
__global__ __launch_bounds__(256)
void conv2_gn_relu(const float* __restrict__ x, const float* __restrict__ w,
                   const float* __restrict__ bias, const float* __restrict__ gg,
                   const float* __restrict__ gb, float* __restrict__ out)
{
    __shared__ float sin[16 * 225];
    __shared__ float sout[32 * 64];
    __shared__ float red[256];
    __shared__ float s_mean, s_rstd;

    const int b = blockIdx.x;
    const float* xin = x + (size_t)b * (16 * 225);
    for (int i = threadIdx.x; i < 16 * 225; i += 256) sin[i] = xin[i];
    __syncthreads();

    for (int o = threadIdx.x; o < 32 * 64; o += 256) {
        int c = o / 64, p = o % 64, oy = p / 8, ox = p % 8;
        float acc = bias[c];
        for (int ic = 0; ic < 16; ++ic) {
            const float* wc = w + ((c * 16 + ic) * 9);
            const float* si = sin + ic * 225;
#pragma unroll
            for (int ky = 0; ky < 3; ++ky) {
                int iy = oy * 2 + ky - 1;
                if (iy < 0 || iy >= 15) continue;
#pragma unroll
                for (int kx = 0; kx < 3; ++kx) {
                    int ix = ox * 2 + kx - 1;
                    if (ix < 0 || ix >= 15) continue;
                    acc += si[iy * 15 + ix] * wc[ky * 3 + kx];
                }
            }
        }
        sout[o] = acc;
    }
    __syncthreads();

    // GroupNorm: 8 groups of 4 channels * 64 = 256 elements
    for (int g = 0; g < 8; ++g) {
        float v0 = sout[g * 256 + threadIdx.x];
        float ts = block_reduce_sum(v0, red);
        if (threadIdx.x == 0) s_mean = ts * (1.f / 256.f);
        __syncthreads();
        float d = v0 - s_mean;
        float ts2 = block_reduce_sum(d * d, red);
        if (threadIdx.x == 0) s_rstd = rsqrtf(ts2 * (1.f / 256.f) + EPS);
        __syncthreads();
        int c = g * 4 + threadIdx.x / 64;
        float v = d * s_rstd * gg[c] + gb[c];
        out[(size_t)b * 2048 + g * 256 + threadIdx.x] = fmaxf(v, 0.f);
        __syncthreads();
    }
}

// ---------------------------------------------------------------------------
// Conv3: 32->64, 3x3, stride 2, pad 1, 8x8 -> 4x4, + GroupNorm(8) + ReLU
// ---------------------------------------------------------------------------
__global__ __launch_bounds__(256)
void conv3_gn_relu(const float* __restrict__ x, const float* __restrict__ w,
                   const float* __restrict__ bias, const float* __restrict__ gg,
                   const float* __restrict__ gb, float* __restrict__ out)
{
    __shared__ float sin[32 * 64];
    __shared__ float sout[64 * 16];
    __shared__ float red[256];
    __shared__ float s_mean, s_rstd;

    const int b = blockIdx.x;
    const float* xin = x + (size_t)b * (32 * 64);
    for (int i = threadIdx.x; i < 32 * 64; i += 256) sin[i] = xin[i];
    __syncthreads();

    for (int o = threadIdx.x; o < 64 * 16; o += 256) {
        int c = o / 16, p = o % 16, oy = p / 4, ox = p % 4;
        float acc = bias[c];
        for (int ic = 0; ic < 32; ++ic) {
            const float* wc = w + ((c * 32 + ic) * 9);
            const float* si = sin + ic * 64;
#pragma unroll
            for (int ky = 0; ky < 3; ++ky) {
                int iy = oy * 2 + ky - 1;
                if (iy < 0 || iy >= 8) continue;
#pragma unroll
                for (int kx = 0; kx < 3; ++kx) {
                    int ix = ox * 2 + kx - 1;
                    if (ix < 0 || ix >= 8) continue;
                    acc += si[iy * 8 + ix] * wc[ky * 3 + kx];
                }
            }
        }
        sout[o] = acc;
    }
    __syncthreads();

    // GroupNorm: 8 groups of 8 channels * 16 = 128 elements
    for (int g = 0; g < 8; ++g) {
        float v0 = (threadIdx.x < 128) ? sout[g * 128 + threadIdx.x] : 0.f;
        float ts = block_reduce_sum(v0, red);
        if (threadIdx.x == 0) s_mean = ts * (1.f / 128.f);
        __syncthreads();
        float d = v0 - s_mean;
        float dd = (threadIdx.x < 128) ? d * d : 0.f;
        float ts2 = block_reduce_sum(dd, red);
        if (threadIdx.x == 0) s_rstd = rsqrtf(ts2 * (1.f / 128.f) + EPS);
        __syncthreads();
        if (threadIdx.x < 128) {
            int c = g * 8 + threadIdx.x / 16;
            float v = d * s_rstd * gg[c] + gb[c];
            out[(size_t)b * 1024 + g * 128 + threadIdx.x] = fmaxf(v, 0.f);
        }
        __syncthreads();
    }
}

// ---------------------------------------------------------------------------
// LayerNorm over 256 features (block-per-row). Optional f32 out and/or f16
// out into a strided concat buffer; optional ReLU.
// ---------------------------------------------------------------------------
__global__ __launch_bounds__(256)
void layernorm256(const float* __restrict__ X, const float* __restrict__ g,
                  const float* __restrict__ b,
                  float* __restrict__ Yf,
                  _Float16* __restrict__ Yh, int ldyh, int ycol,
                  int relu)
{
    __shared__ float red[256];
    __shared__ float sm, srs;
    const int row = blockIdx.x, t = threadIdx.x;
    float v = X[(size_t)row * 256 + t];
    float ts = block_reduce_sum(v, red);
    if (t == 0) sm = ts * (1.f / 256.f);
    __syncthreads();
    float d = v - sm;
    float ts2 = block_reduce_sum(d * d, red);
    if (t == 0) srs = rsqrtf(ts2 * (1.f / 256.f) + EPS);
    __syncthreads();
    float y = d * srs * g[t] + b[t];
    if (relu) y = fmaxf(y, 0.f);
    if (Yf) Yf[(size_t)row * 256 + t] = y;
    if (Yh) Yh[(size_t)row * ldyh + ycol + t] = (_Float16)y;
}

// ---------------------------------------------------------------------------
// comm projection: (B*8, 4) @ (256,4)^T + b -> f16 (B*8, 256). K=4: VALU.
// ---------------------------------------------------------------------------
__global__ void comm_proj_kernel(const float* __restrict__ comm, const float* __restrict__ w,
                                 const float* __restrict__ bias, _Float16* __restrict__ out,
                                 long long total)
{
    long long idx = (long long)blockIdx.x * 256 + threadIdx.x;
    if (idx >= total) return;
    long long row = idx >> 8;
    int c = (int)(idx & 255);
    const float* ci = comm + row * 4;
    float acc = bias[c];
#pragma unroll
    for (int j = 0; j < 4; ++j) acc += ci[j] * w[c * 4 + j];
    out[idx] = (_Float16)acc;
}

// ---------------------------------------------------------------------------
// scores -> softmax(8) -> context, block-per-sample (t = feature dim)
// ---------------------------------------------------------------------------
__global__ __launch_bounds__(256)
void attn_kernel(const float* __restrict__ Q, const _Float16* __restrict__ Kp,
                 const _Float16* __restrict__ Vp, _Float16* __restrict__ ctx0)
{
    __shared__ float red[256];
    __shared__ float sc[8];
    const int b = blockIdx.x, t = threadIdx.x;
    const float q = Q[(size_t)b * 256 + t];
    for (int k = 0; k < 8; ++k) {
        float p = q * (float)Kp[((size_t)b * 8 + k) * 256 + t];
        float ts = block_reduce_sum(p, red);
        if (t == 0) sc[k] = ts * 0.0625f;  // 1/sqrt(256)
        __syncthreads();
    }
    float mx = sc[0];
#pragma unroll
    for (int k = 1; k < 8; ++k) mx = fmaxf(mx, sc[k]);
    float wk[8], sum = 0.f;
#pragma unroll
    for (int k = 0; k < 8; ++k) { wk[k] = expf(sc[k] - mx); sum += wk[k]; }
    const float inv = 1.f / sum;
    float acc = 0.f;
#pragma unroll
    for (int k = 0; k < 8; ++k)
        acc += wk[k] * inv * (float)Vp[((size_t)b * 8 + k) * 256 + t];
    ctx0[(size_t)b * 256 + t] = (_Float16)acc;
}

// ---------------------------------------------------------------------------
// LSTM pointwise (gate order i,f,g,o), writes h_new and c_new
// ---------------------------------------------------------------------------
__global__ void lstm_pointwise(const float* __restrict__ gates, const float* __restrict__ c0,
                               float* __restrict__ hout, float* __restrict__ cout,
                               long long total)
{
    long long idx = (long long)blockIdx.x * 256 + threadIdx.x;
    if (idx >= total) return;
    long long b = idx >> 8;
    int d = (int)(idx & 255);
    const float* g = gates + b * 1024;
    float i = sigmoidf_(g[d]);
    float f = sigmoidf_(g[256 + d]);
    float gg = tanhf(g[512 + d]);
    float o = sigmoidf_(g[768 + d]);
    float c = f * c0[idx] + i * gg;
    float h = o * tanhf(c);
    cout[idx] = c;
    hout[idx] = h;
}

// ---------------------------------------------------------------------------
// Heads: logits(12), tanh mu(4), clipped logstd(4) from feat(256)
// ---------------------------------------------------------------------------
__global__ __launch_bounds__(256)
void heads_kernel(const float* __restrict__ feat,
                  const float* __restrict__ wa, const float* __restrict__ ba,
                  const float* __restrict__ wm, const float* __restrict__ bm,
                  const float* __restrict__ wl, const float* __restrict__ bl,
                  float* __restrict__ logits, float* __restrict__ mu,
                  float* __restrict__ logstd)
{
    __shared__ float sf[256];
    const int b = blockIdx.x, t = threadIdx.x;
    sf[t] = feat[(size_t)b * 256 + t];
    __syncthreads();
    if (t < 12) {
        float acc = ba[t];
        for (int j = 0; j < 256; ++j) acc += wa[t * 256 + j] * sf[j];
        logits[(size_t)b * 12 + t] = acc;
    } else if (t < 16) {
        int r = t - 12;
        float acc = bm[r];
        for (int j = 0; j < 256; ++j) acc += wm[r * 256 + j] * sf[j];
        mu[(size_t)b * 4 + r] = tanhf(acc);
    } else if (t < 20) {
        int r = t - 16;
        float acc = bl[r];
        for (int j = 0; j < 256; ++j) acc += wl[r * 256 + j] * sf[j];
        logstd[(size_t)b * 4 + r] = fminf(fmaxf(acc, -4.f), 0.f);
    }
}

// ---------------------------------------------------------------------------
// Host-side orchestration
// ---------------------------------------------------------------------------
extern "C" void kernel_launch(void* const* d_in, const int* in_sizes, int n_in,
                              void* d_out, int out_size, void* d_ws, size_t ws_size,
                              hipStream_t stream)
{
    const float* x        = (const float*)d_in[0];
    const float* scalars  = (const float*)d_in[1];
    const float* h0       = (const float*)d_in[2];
    const float* c0       = (const float*)d_in[3];
    const float* comm_in  = (const float*)d_in[4];
    const float* conv1_w  = (const float*)d_in[5];
    const float* conv1_b  = (const float*)d_in[6];
    const float* gn1_g    = (const float*)d_in[7];
    const float* gn1_b    = (const float*)d_in[8];
    const float* conv2_w  = (const float*)d_in[9];
    const float* conv2_b  = (const float*)d_in[10];
    const float* gn2_g    = (const float*)d_in[11];
    const float* gn2_b    = (const float*)d_in[12];
    const float* conv3_w  = (const float*)d_in[13];
    const float* conv3_b  = (const float*)d_in[14];
    const float* gn3_g    = (const float*)d_in[15];
    const float* gn3_b    = (const float*)d_in[16];
    const float* fc_embed_w = (const float*)d_in[17];
    const float* fc_embed_b = (const float*)d_in[18];
    const float* en_g     = (const float*)d_in[19];
    const float* en_b     = (const float*)d_in[20];
    const float* comm_w   = (const float*)d_in[21];
    const float* comm_b   = (const float*)d_in[22];
    const float* q_w      = (const float*)d_in[23];
    const float* q_b      = (const float*)d_in[24];
    const float* k_w      = (const float*)d_in[25];
    const float* k_b      = (const float*)d_in[26];
    const float* v_w      = (const float*)d_in[27];
    const float* v_b      = (const float*)d_in[28];
    const float* o_w      = (const float*)d_in[29];
    const float* o_b      = (const float*)d_in[30];
    const float* fuse_w   = (const float*)d_in[31];
    const float* fuse_b   = (const float*)d_in[32];
    const float* fn_g     = (const float*)d_in[33];
    const float* fn_b     = (const float*)d_in[34];
    const float* w_ih     = (const float*)d_in[35];
    const float* w_hh     = (const float*)d_in[36];
    const float* b_ih     = (const float*)d_in[37];
    const float* b_hh     = (const float*)d_in[38];
    const float* ln_g     = (const float*)d_in[39];
    const float* ln_b     = (const float*)d_in[40];
    const float* wa       = (const float*)d_in[41];
    const float* ba       = (const float*)d_in[42];
    const float* wm       = (const float*)d_in[43];
    const float* bm       = (const float*)d_in[44];
    const float* wl       = (const float*)d_in[45];
    const float* bl       = (const float*)d_in[46];

    const size_t B = (size_t)in_sizes[0] / (6 * 225);   // 16384

    // --- workspace overlay (manually aliased; peak ~373 MB for B=16384) ---
    char* ws = (char*)d_ws;
    const size_t oC1 = 0;                               // f32 B*3600 (conv1 out)
    const size_t oC2 = B * 3600 * 4;                    // f32 B*2048 (conv2 out)
    const size_t oC3 = 0;                               // f32 B*1024 (reuses c1)
    const size_t oXE = B * 1024 * 4;                    // f16 B*1056 (embed input)
    const size_t oEM = oXE + B * 1056 * 2;              // f32 B*256  (embed pre-LN)
    const size_t oH0 = oEM + B * 256 * 4;               // f16 B*256  (h0 as f16)
    const size_t oQ  = oH0 + B * 256 * 2;               // f32 B*256  (Q)
    const size_t oKV = oQ + B * 256 * 4;                // f16 B*2048 (KV proj)
    const size_t oKP = oKV + B * 2048 * 2;              // f16 B*2048 (K')
    const size_t oVP = oKP + B * 2048 * 2;              // f16 B*2048 (V')
    const size_t oCX = oXE;                             // f16 B*256  (ctx0; Xe dead)
    const size_t oX2 = oCX + B * 256 * 2;               // f16 B*512  ([embed|ctx])
    const size_t oFU = oX2 + B * 512 * 2;               // f32 B*256  (fused pre-LN)
    const size_t oX3 = oFU + B * 256 * 4;               // f16 B*512  ([fused|h0])
    const size_t oGA = oKV;                             // f32 B*1024 (gates; KV dead)
    const size_t oWP = oC2 + B * 2048 * 4;              // packed weights
    const size_t oWE = oWP;                             // f16 256*1056
    const size_t oWQ = oWE + 256 * 1056 * 2;
    const size_t oWK = oWQ + 256 * 256 * 2;
    const size_t oWV = oWK + 256 * 256 * 2;
    const size_t oWO = oWV + 256 * 256 * 2;
    const size_t oWF = oWO + 256 * 256 * 2;             // f16 256*512
    const size_t oWG = oWF + 256 * 512 * 2;             // f16 1024*512
    const size_t oBG = oWG + 1024 * 512 * 2;            // f32 1024

    float*     c1    = (float*)(ws + oC1);
    float*     c2    = (float*)(ws + oC2);
    float*     c3    = (float*)(ws + oC3);
    _Float16*  Xe    = (_Float16*)(ws + oXE);
    float*     embed = (float*)(ws + oEM);
    _Float16*  h0h   = (_Float16*)(ws + oH0);
    float*     Qb    = (float*)(ws + oQ);
    _Float16*  KV    = (_Float16*)(ws + oKV);
    _Float16*  KPb   = (_Float16*)(ws + oKP);
    _Float16*  VPb   = (_Float16*)(ws + oVP);
    _Float16*  ctx0  = (_Float16*)(ws + oCX);
    _Float16*  X2    = (_Float16*)(ws + oX2);
    float*     fused = (float*)(ws + oFU);
    _Float16*  X3    = (_Float16*)(ws + oX3);
    float*     gates = (float*)(ws + oGA);
    _Float16*  WE    = (_Float16*)(ws + oWE);
    _Float16*  WQ    = (_Float16*)(ws + oWQ);
    _Float16*  WK    = (_Float16*)(ws + oWK);
    _Float16*  WV    = (_Float16*)(ws + oWV);
    _Float16*  WO    = (_Float16*)(ws + oWO);
    _Float16*  WF    = (_Float16*)(ws + oWF);
    _Float16*  WG    = (_Float16*)(ws + oWG);
    float*     BG    = (float*)(ws + oBG);

    // d_out layout: logits(12) | mu(4) | logstd(4) | feat(256) | h(256) | c(256)
    float* out       = (float*)d_out;
    float* o_logits  = out;
    float* o_mu      = out + B * 12;
    float* o_logstd  = out + B * 16;
    float* o_feat    = out + B * 20;
    float* o_h       = out + B * 276;
    float* o_c       = out + B * 532;

    // --- weight packing (cheap; deterministic every call) ---
    pack_w_f16<<<(256 * 1056 + 255) / 256, 256, 0, stream>>>(fc_embed_w, WE, 256, 1046, 1056);
    pack_w_f16<<<(256 * 256 + 255) / 256, 256, 0, stream>>>(q_w, WQ, 256, 256, 256);
    pack_w_f16<<<(256 * 256 + 255) / 256, 256, 0, stream>>>(k_w, WK, 256, 256, 256);
    pack_w_f16<<<(256 * 256 + 255) / 256, 256, 0, stream>>>(v_w, WV, 256, 256, 256);
    pack_w_f16<<<(256 * 256 + 255) / 256, 256, 0, stream>>>(o_w, WO, 256, 256, 256);
    pack_w_f16<<<(256 * 512 + 255) / 256, 256, 0, stream>>>(fuse_w, WF, 256, 512, 512);
    pack_gates<<<(1024 * 512 + 255) / 256, 256, 0, stream>>>(w_ih, w_hh, b_ih, b_hh, WG, BG);

    // --- CNN trunk (fused conv + GN + ReLU) ---
    conv1_gn_relu<<<(unsigned)B, 256, 0, stream>>>(x, conv1_w, conv1_b, gn1_g, gn1_b, c1);
    conv2_gn_relu<<<(unsigned)B, 256, 0, stream>>>(c1, conv2_w, conv2_b, gn2_g, gn2_b, c2);
    conv3_gn_relu<<<(unsigned)B, 256, 0, stream>>>(c2, conv3_w, conv3_b, gn3_g, gn3_b, c3);

    // --- assemble embed input [flat(1024) | scalars(22) | zeros(10)] as f16 ---
    {
        long long t1 = (long long)B * 1024;
        f32_to_f16_strided<<<(unsigned)((t1 + 255) / 256), 256, 0, stream>>>(c3, Xe, t1, 1024, 1056, 0);
        long long t2 = (long long)B * 22;
        f32_to_f16_strided<<<(unsigned)((t2 + 255) / 256), 256, 0, stream>>>(scalars, Xe, t2, 22, 1056, 1024);
        long long t3 = (long long)B * 10;
        f16_fill_zero<<<(unsigned)((t3 + 255) / 256), 256, 0, stream>>>(Xe, t3, 10, 1056, 1046);
    }

    // --- fc_embed (WMMA) + LN + ReLU (f16 result into X2 cols [0,256)) ---
    gemm_f16_wmma<<<dim3((unsigned)(B / 64), 1), 256, 0, stream>>>(
        Xe, 1056, WE, fc_embed_b, embed, 256, ((_Float16*)nullptr), 0, 0, 1056);
    layernorm256<<<(unsigned)B, 256, 0, stream>>>(embed, en_g, en_b,
        (float*)nullptr, X2, 512, 0, 1);

    // --- attention ---
    {
        long long th0 = (long long)B * 256;
        f32_to_f16_strided<<<(unsigned)((th0 + 255) / 256), 256, 0, stream>>>(h0, h0h, th0, 256, 256, 0);
    }
    gemm_f16_wmma<<<dim3((unsigned)(B / 64), 1), 256, 0, stream>>>(
        h0h, 256, WQ, q_b, Qb, 256, ((_Float16*)nullptr), 0, 0, 256);

    {
        long long tc = (long long)B * 8 * 256;
        comm_proj_kernel<<<(unsigned)((tc + 255) / 256), 256, 0, stream>>>(comm_in, comm_w, comm_b, KV, tc);
    }
    gemm_f16_wmma<<<dim3((unsigned)(B * 8 / 64), 1), 256, 0, stream>>>(
        KV, 256, WK, k_b, (float*)nullptr, 0, KPb, 256, 0, 256);
    gemm_f16_wmma<<<dim3((unsigned)(B * 8 / 64), 1), 256, 0, stream>>>(
        KV, 256, WV, v_b, (float*)nullptr, 0, VPb, 256, 0, 256);

    attn_kernel<<<(unsigned)B, 256, 0, stream>>>(Qb, KPb, VPb, ctx0);

    // ctx = attn_out(ctx0): f16 straight into X2 cols [256,512)
    gemm_f16_wmma<<<dim3((unsigned)(B / 64), 1), 256, 0, stream>>>(
        ctx0, 256, WO, o_b, (float*)nullptr, 0, X2, 512, 256, 256);

    // --- fc_fuse (WMMA) + LN + ReLU (f16 into X3 cols [0,256)) ---
    gemm_f16_wmma<<<dim3((unsigned)(B / 64), 1), 256, 0, stream>>>(
        X2, 512, WF, fuse_b, fused, 256, ((_Float16*)nullptr), 0, 0, 512);
    layernorm256<<<(unsigned)B, 256, 0, stream>>>(fused, fn_g, fn_b,
        (float*)nullptr, X3, 512, 0, 1);

    // X3 cols [256,512) = h0 (f16)
    {
        long long th0 = (long long)B * 256;
        f32_to_f16_strided<<<(unsigned)((th0 + 255) / 256), 256, 0, stream>>>(h0, X3, th0, 256, 512, 256);
    }

    // --- LSTM gates = [fused|h0] @ [w_ih|w_hh]^T + (b_ih + b_hh) ---
    gemm_f16_wmma<<<dim3((unsigned)(B / 64), 4), 256, 0, stream>>>(
        X3, 512, WG, BG, gates, 1024, ((_Float16*)nullptr), 0, 0, 512);

    {
        long long tl = (long long)B * 256;
        lstm_pointwise<<<(unsigned)((tl + 255) / 256), 256, 0, stream>>>(gates, c0, o_h, o_c, tl);
    }

    // feat = LN(h_new)
    layernorm256<<<(unsigned)B, 256, 0, stream>>>(o_h, ln_g, ln_b,
        o_feat, (_Float16*)nullptr, 0, 0, 0);

    // heads
    heads_kernel<<<(unsigned)B, 256, 0, stream>>>(o_feat, wa, ba, wm, bm, wl, bl,
                                                  o_logits, o_mu, o_logstd);
}
